// SparseMultiRangeCrossAttention_16733192585755
// MI455X (gfx1250) — matrix operand order repaired and checked by reading, not compiled, not run on hardware
//
#include <hip/hip_runtime.h>
#include <hip/hip_bf16.h>

typedef __bf16 bf16_t;
typedef __attribute__((ext_vector_type(16))) __bf16 v16bf;
typedef __attribute__((ext_vector_type(8)))  float  v8f;

#define WMMA_BF16(a, b, c) \
  __builtin_amdgcn_wmma_f32_16x16x32_bf16(false, (a), false, (b), (short)0, (c), false, false)

union FragU { uint4 u[2]; v16bf v; };

__device__ __forceinline__ v16bf frag2(const bf16_t* p0, const bf16_t* p1) {
  FragU f;
  f.u[0] = *(const uint4*)p0;
  f.u[1] = *(const uint4*)p1;
  return f.v;
}
__device__ __forceinline__ v16bf frag32B(const bf16_t* p) {
  FragU f;
  f.u[0] = *(const uint4*)p;
  f.u[1] = *(const uint4*)(p + 8);
  return f.v;
}
__device__ __forceinline__ v8f v8zero() {
  v8f z = {0.f, 0.f, 0.f, 0.f, 0.f, 0.f, 0.f, 0.f};
  return z;
}

// gfx1250 async global->LDS copy (ASYNCcnt-tracked), 16B per lane.
// lds_off = byte offset within the workgroup LDS allocation (low 32 bits of a
// flat shared pointer are the LDS offset per CDNA5 aperture rules).
__device__ __forceinline__ void async_ld_b128(unsigned lds_off, const void* gptr) {
  asm volatile("global_load_async_to_lds_b128 %0, %1, off"
               :: "v"(lds_off), "v"((unsigned long long)(size_t)gptr)
               : "memory");
}
__device__ __forceinline__ void wait_async0() {
  asm volatile("s_wait_asynccnt 0x0" ::: "memory");
}
__device__ __forceinline__ void wait_ds0() {
  asm volatile("s_wait_dscnt 0x0" ::: "memory");
}

#define BATCH 8
#define HH 128
#define WW 128
#define CPROJ 192
#define SCALE_Q 0.17677669529663687f  // 32^-0.5

// ---------------------------------------------------------------------------
// Kernel T: weights f32 [K][N] -> bf16 transposed [N][K] (contiguous B-frags).
// ---------------------------------------------------------------------------
__global__ void k_transpose_w(const float* __restrict__ Wq,
                              const float* __restrict__ Wk,
                              const float* __restrict__ Wv,
                              const float* __restrict__ Wo,
                              bf16_t* __restrict__ WT) {
  int idx = blockIdx.x * 256 + threadIdx.x;
  if (idx >= 4 * 192 * 192) return;
  int mat = idx / (192 * 192);
  int r = idx % (192 * 192);
  int k = r / 192, n = r % 192;
  const float* src = (mat == 0) ? Wq : (mat == 1) ? Wk : (mat == 2) ? Wv : Wo;
  WT[(size_t)mat * 192 * 192 + (size_t)n * 192 + k] = (bf16_t)src[(size_t)k * 192 + n];
}

// ---------------------------------------------------------------------------
// Kernel A: fused Q/K/V projection. One workgroup (8 waves) per (b,h) row.
// ---------------------------------------------------------------------------
__global__ __launch_bounds__(256) void k_qkv(
    const float* __restrict__ x, const float* __restrict__ ctx,
    const bf16_t* __restrict__ WT,
    const float* __restrict__ bq, const float* __restrict__ bk,
    const float* __restrict__ bv,
    bf16_t* __restrict__ Q, bf16_t* __restrict__ Kt, bf16_t* __restrict__ Vt) {
  __shared__ __align__(16) bf16_t S[128][200];  // [pixel w][channel], 400B rows

  int bh = blockIdx.x;
  int b = bh >> 7, h = bh & 127;
  int tid = threadIdx.x;
  int wave = tid >> 5, lane = tid & 31;
  int grp = lane >> 4, ln = lane & 15;
  int mb = wave * 16;

  for (int mat = 0; mat < 3; ++mat) {
    if (mat < 2) {
      if (mat == 1) __syncthreads();  // all waves done reading x-tile
      const float* src = (mat == 0) ? x : ctx;
      // float4-vectorized staged transpose: [c][w] global -> [w][c] LDS bf16
      for (int i = tid; i < 192 * 32; i += 256) {
        int c = i >> 5, w4 = (i & 31) << 2;
        const float4 f =
            *(const float4*)&src[(((size_t)b * 192 + c) * 128 + h) * 128 + w4];
        S[w4 + 0][c] = (bf16_t)f.x;
        S[w4 + 1][c] = (bf16_t)f.y;
        S[w4 + 2][c] = (bf16_t)f.z;
        S[w4 + 3][c] = (bf16_t)f.w;
      }
      __syncthreads();
    }
    const bf16_t* Wm = WT + (size_t)mat * 192 * 192;
    const float* bias = (mat == 0) ? bq : (mat == 1) ? bk : bv;
    bf16_t* dst = (mat == 0) ? Q : (mat == 1) ? Kt : Vt;

    v8f acc[12];
#pragma unroll
    for (int nf = 0; nf < 12; ++nf) acc[nf] = v8zero();

    for (int kb = 0; kb < 6; ++kb) {
      v16bf a = frag2(&S[mb + ln][kb * 32 + grp * 16],
                      &S[mb + ln][kb * 32 + grp * 16 + 8]);
#pragma unroll
      for (int nf = 0; nf < 12; ++nf) {
        v16bf bw = frag32B(Wm + (size_t)(nf * 16 + ln) * 192 + kb * 32 + grp * 16);
        acc[nf] = WMMA_BF16(a, bw, acc[nf]);
      }
    }
    float sc = (mat == 0) ? SCALE_Q : 1.0f;
#pragma unroll
    for (int nf = 0; nf < 12; ++nf) {
      int n = nf * 16 + ln;
      float bn = bias[n];
#pragma unroll
      for (int r = 0; r < 8; ++r) {
        int w = mb + r + grp * 8;
        size_t pix = ((size_t)b * 128 + h) * 128 + w;
        dst[pix * 192 + n] = (bf16_t)((acc[nf][r] + bn) * sc);
      }
    }
  }
}

// ---------------------------------------------------------------------------
// Kernel B: neighborhood attention, flash-style online softmax.
// One wave per (b, h, 16-query tile, branch, head). Per window row:
//   async-DMA the 32x32 bf16 K-slab into LDS, 2 WMMAs QK^T, j-invariant
//   mask/RPB-column hoisted, online softmax (16-lane shuffles), P/V relaid
//   through LDS, 2 WMMAs P*V.
// ---------------------------------------------------------------------------
__global__ __launch_bounds__(32) void k_natten(
    const bf16_t* __restrict__ Q, const bf16_t* __restrict__ Kt,
    const bf16_t* __restrict__ Vt,
    const float* __restrict__ rpb0, const float* __restrict__ rpb1,
    const float* __restrict__ rpb2,
    bf16_t* __restrict__ AO) {
  __shared__ __align__(16) bf16_t Pl[16][40];  // probs: [query][slab pos]
  __shared__ __align__(16) bf16_t Vl[32][40];  // V transposed: [dim][slab pos]
  __shared__ __align__(16) bf16_t Kl[32][40];  // K slab: [slab pos][dim]

  int wt = blockIdx.x / 6;
  int bh6 = blockIdx.x % 6;
  int branch = bh6 >> 1, head = bh6 & 1;
  int h = blockIdx.y, b = blockIdx.z;

  int K = (branch == 0) ? 7 : (branch == 1) ? 9 : 11;
  int NS = K >> 1;
  int Km1 = K - 1, D = 2 * K - 1;
  const float* rpb =
      ((branch == 0) ? rpb0 : (branch == 1) ? rpb1 : rpb2) + (size_t)head * D * D;
  int c0 = branch * 64 + head * 32;
  int w0 = wt * 16;

  int lane = threadIdx.x;
  int grp = lane >> 4, ln = lane & 15;

  // Q A-fragment (pre-scaled at projection time)
  size_t qpix = ((size_t)b * 128 + h) * 128 + (w0 + ln);
  const bf16_t* qrow = Q + qpix * 192 + c0;
  v16bf qa = frag2(qrow + grp * 16, qrow + grp * 16 + 8);

  int hs = h - NS; if (hs < 0) hs = 0; if (hs > HH - K) hs = HH - K;
  int cb = w0 - NS; if (cb < 0) cb = 0; if (cb > WW - K) cb = WW - K;
  if (cb > WW - 32) cb = WW - 32;  // 32-wide slab in-bounds, covers all windows

  // j-invariant per-element window masks and RPB column offsets
  bool va0[8], va1[8];
  int bc0[8], bc1[8];
#pragma unroll
  for (int r = 0; r < 8; ++r) {
    int wq = w0 + r + grp * 8;
    int cs = wq - NS; if (cs < 0) cs = 0; if (cs > WW - K) cs = WW - K;
    int kc0 = cb + ln;
    int kc1 = kc0 + 16;
    va0[r] = (kc0 >= cs) && (kc0 < cs + K);
    va1[r] = (kc1 >= cs) && (kc1 < cs + K);
    bc0[r] = kc0 - wq + Km1;
    bc1[r] = kc1 - wq + Km1;
  }

  float mo[8], ls[8];
#pragma unroll
  for (int r = 0; r < 8; ++r) { mo[r] = -1e30f; ls[r] = 0.f; }
  v8f acc0 = v8zero(), acc1 = v8zero();

  unsigned kl_base = (unsigned)(unsigned long long)(size_t)&Kl[0][0];

  for (int j = 0; j < K; ++j) {
    int hr = hs + j;
    size_t krow = ((size_t)b * 128 + hr) * 128;

    // Async DMA K slab [32 pos][32 dims] bf16 into LDS (lane owns one pos).
    {
      const bf16_t* kg = Kt + (krow + cb + lane) * 192 + c0;
      unsigned loff = kl_base + (unsigned)(lane * 40) * 2u;
#pragma unroll
      for (int ch = 0; ch < 4; ++ch)
        async_ld_b128(loff + ch * 16u, kg + ch * 8);
    }
    // Prefetch next window row's K and V lines while this row computes.
    if (j + 1 < K) {
      size_t nrow = ((size_t)b * 128 + hr + 1) * 128;
      __builtin_prefetch(Kt + (nrow + cb + lane) * 192 + c0, 0, 3);
      __builtin_prefetch(Vt + (nrow + cb + lane) * 192 + c0, 0, 3);
    }

    // Stage V slab transposed: Vl[dim][pos]; lane owns pos = cb + lane
    {
      const bf16_t* vp = Vt + (krow + cb + lane) * 192 + c0;
      union { uint4 u; bf16_t e[8]; } u0, u1, u2, u3;
      u0.u = *(const uint4*)(vp);
      u1.u = *(const uint4*)(vp + 8);
      u2.u = *(const uint4*)(vp + 16);
      u3.u = *(const uint4*)(vp + 24);
#pragma unroll
      for (int d = 0; d < 8; ++d) {
        Vl[d][lane] = u0.e[d];
        Vl[8 + d][lane] = u1.e[d];
        Vl[16 + d][lane] = u2.e[d];
        Vl[24 + d][lane] = u3.e[d];
      }
    }

    wait_async0();  // K slab resident in LDS
    v16bf kb0 = frag32B(&Kl[ln][grp * 16]);
    v16bf kb1 = frag32B(&Kl[16 + ln][grp * 16]);
    v8f s0 = WMMA_BF16(qa, kb0, v8zero());
    v8f s1 = WMMA_BF16(qa, kb1, v8zero());

    int bias_r = hs + j - h + Km1;
    const float* rpbr = rpb + (size_t)bias_r * D;

#pragma unroll
    for (int r = 0; r < 8; ++r) {
      float t0 = va0[r] ? (s0[r] + rpbr[bc0[r]]) : -1e30f;
      float t1 = va1[r] ? (s1[r] + rpbr[bc1[r]]) : -1e30f;

      float mr = fmaxf(t0, t1);
      mr = fmaxf(mr, __shfl_xor(mr, 1));
      mr = fmaxf(mr, __shfl_xor(mr, 2));
      mr = fmaxf(mr, __shfl_xor(mr, 4));
      mr = fmaxf(mr, __shfl_xor(mr, 8));

      float mn = fmaxf(mo[r], mr);
      float scl = __expf(mo[r] - mn);
      mo[r] = mn;

      float p0 = __expf(t0 - mn);
      float p1 = __expf(t1 - mn);
      float ps = p0 + p1;
      ps += __shfl_xor(ps, 1);
      ps += __shfl_xor(ps, 2);
      ps += __shfl_xor(ps, 4);
      ps += __shfl_xor(ps, 8);
      ls[r] = ls[r] * scl + ps;
      acc0[r] *= scl;
      acc1[r] *= scl;

      Pl[r + grp * 8][ln] = (bf16_t)p0;
      Pl[r + grp * 8][16 + ln] = (bf16_t)p1;
    }
    __syncthreads();

    v16bf pa = frag2(&Pl[ln][grp * 16], &Pl[ln][grp * 16 + 8]);
    v16bf vb0 = frag32B(&Vl[ln][grp * 16]);       // out dims c0..c0+15
    v16bf vb1 = frag32B(&Vl[16 + ln][grp * 16]);  // out dims c0+16..c0+31
    acc0 = WMMA_BF16(pa, vb0, acc0);
    acc1 = WMMA_BF16(pa, vb1, acc1);

    wait_ds0();  // LDS reads drained before next iteration's async/ds writes
    __syncthreads();
  }

#pragma unroll
  for (int r = 0; r < 8; ++r) {
    float inv = 1.0f / ls[r];
    int w = w0 + r + grp * 8;
    size_t pix = ((size_t)b * 128 + h) * 128 + w;
    AO[pix * 192 + c0 + ln] = (bf16_t)(acc0[r] * inv);
    AO[pix * 192 + c0 + 16 + ln] = (bf16_t)(acc1[r] * inv);
  }
}

// ---------------------------------------------------------------------------
// Kernel C: output projection AO[131072x192] @ Wo + bo -> fp32 [B,192,H,W].
// C-fragments are transposed through LDS (two 96-channel phases) so the
// channel-major fp32 output is written as fully coalesced float4 rows.
// ---------------------------------------------------------------------------
__global__ __launch_bounds__(256) void k_oproj(
    const bf16_t* __restrict__ AO, const bf16_t* __restrict__ WT,
    const float* __restrict__ bo, float* __restrict__ out) {
  __shared__ __align__(16) float O[96][132];  // [channel][w], padded rows

  int bh = blockIdx.x;
  int b = bh >> 7, h = bh & 127;
  int tid = threadIdx.x;
  int wave = tid >> 5, lane = tid & 31;
  int grp = lane >> 4, ln = lane & 15;
  int w0 = wave * 16;

  const bf16_t* WoT = WT + (size_t)3 * 192 * 192;
  size_t pixbase = ((size_t)b * 128 + h) * 128 + w0;
  const bf16_t* arow = AO + (pixbase + ln) * 192;

  v8f acc[12];
#pragma unroll
  for (int nf = 0; nf < 12; ++nf) acc[nf] = v8zero();

  for (int kb = 0; kb < 6; ++kb) {
    v16bf a = frag2(arow + kb * 32 + grp * 16, arow + kb * 32 + grp * 16 + 8);
#pragma unroll
    for (int nf = 0; nf < 12; ++nf) {
      v16bf bw = frag32B(WoT + (size_t)(nf * 16 + ln) * 192 + kb * 32 + grp * 16);
      acc[nf] = WMMA_BF16(a, bw, acc[nf]);
    }
  }

  for (int p = 0; p < 2; ++p) {
#pragma unroll
    for (int nf = p * 6; nf < p * 6 + 6; ++nf) {
      int n = nf * 16 + ln;
      float bn = bo[n];
#pragma unroll
      for (int r = 0; r < 8; ++r)
        O[n - p * 96][w0 + r + grp * 8] = acc[nf][r] + bn;
    }
    __syncthreads();
    for (int i = tid; i < 96 * 32; i += 256) {
      int nn = i >> 5, w4 = (i & 31) << 2;
      int n = p * 96 + nn;
      float4 v = *(const float4*)&O[nn][w4];
      *(float4*)&out[(((size_t)b * 192 + n) * 128 + h) * 128 + w4] = v;
    }
    __syncthreads();
  }
}

// ---------------------------------------------------------------------------
extern "C" void kernel_launch(void* const* d_in, const int* in_sizes, int n_in,
                              void* d_out, int out_size, void* d_ws, size_t ws_size,
                              hipStream_t stream) {
  (void)in_sizes; (void)n_in; (void)out_size; (void)ws_size;
  const float* x    = (const float*)d_in[0];
  const float* ctx  = (const float*)d_in[1];
  const float* Wq   = (const float*)d_in[2];
  const float* bq   = (const float*)d_in[3];
  const float* Wk   = (const float*)d_in[4];
  const float* bk   = (const float*)d_in[5];
  const float* Wv   = (const float*)d_in[6];
  const float* bv   = (const float*)d_in[7];
  const float* Wo   = (const float*)d_in[8];
  const float* bo   = (const float*)d_in[9];
  const float* rpb0 = (const float*)d_in[10];
  const float* rpb1 = (const float*)d_in[11];
  const float* rpb2 = (const float*)d_in[12];

  char* ws = (char*)d_ws;
  const size_t WT_BYTES = (size_t)4 * 192 * 192 * sizeof(bf16_t);
  const size_t TEN_BYTES = (size_t)BATCH * HH * WW * CPROJ * sizeof(bf16_t);
  bf16_t* WT = (bf16_t*)ws;
  bf16_t* Q  = (bf16_t*)(ws + WT_BYTES);
  bf16_t* Kt = (bf16_t*)(ws + WT_BYTES + TEN_BYTES);
  bf16_t* Vt = (bf16_t*)(ws + WT_BYTES + 2 * TEN_BYTES);
  bf16_t* AO = (bf16_t*)(ws + WT_BYTES + 3 * TEN_BYTES);

  k_transpose_w<<<576, 256, 0, stream>>>(Wq, Wk, Wv, Wo, WT);
  k_qkv<<<BATCH * HH, 256, 0, stream>>>(x, ctx, WT, bq, bk, bv, Q, Kt, Vt);
  k_natten<<<dim3((WW / 16) * 6, HH, BATCH), 32, 0, stream>>>(Q, Kt, Vt, rpb0, rpb1,
                                                              rpb2, AO);
  k_oproj<<<BATCH * HH, 256, 0, stream>>>(AO, WT, bo, (float*)d_out);
}
